// ShenParsingNetwork_73212012527679
// MI455X (gfx1250) — matrix operand order repaired and checked by reading, not compiled, not run on hardware
//
#include <hip/hip_runtime.h>
#include <hip/hip_bf16.h>

typedef __attribute__((ext_vector_type(16))) __bf16 v16bf;
typedef __attribute__((ext_vector_type(8)))  float  v8f;
typedef unsigned int v4u __attribute__((ext_vector_type(4)));
typedef int          v8i __attribute__((ext_vector_type(8)));
typedef int          v4i __attribute__((ext_vector_type(4)));

#define TT    1024
#define BB    64
#define NINP  800
#define NHID  1200
#define KDIM  1600          // 2 * NINP (conv kernel size 2 folded into K)
#define NSLOTS 15
#define MT    128           // t-rows per workgroup (8 waves x 16)
#define AROWS (MT + 1)      // overlapping emb rows needed (k=2 conv)
#define ASTR  808           // halfwords per A row in LDS (800 + 8 pad, 16B aligned)
#define WSTR  1608          // halfwords per W row in LDS (1600 + 8 pad, 16B aligned)
#define NTILE (NHID / 16)   // 75 channel tiles
#define KHALF (NINP / 32)   // 25 k-steps per conv-tap region (800 % 32 == 0)

#define SA_BYTES ((size_t)AROWS * ASTR * 2)      // 208,464
#define SW_BYTES ((size_t)16 * WSTR * 2)         // 51,456
#define LDS_TOTAL (SA_BYTES + 2 * SW_BYTES)      // 311,376 <= 320KB

// output layout (flat concat, fp32):
#define OFF_MG    0
#define OFF_MGN   983040
#define OFF_GATE  1966080
#define OFF_DIST  2031616
#define OFF_EMB   2097152
#define OFF_CUM   2148352

__device__ __forceinline__ unsigned short f2bf(float f) {
  union { __bf16 b; unsigned short u; } cv; cv.b = (__bf16)f; return cv.u;
}
__device__ __forceinline__ float sigmf(float x) { return 1.0f / (1.0f + __expf(-x)); }

// ---- TDM: async-load one 16x1600 bf16 W tile into LDS (padded rows) ----
// D# per CDNA5 ISA. Iteration mode: 16 iterations of a
// 1600x1 row-tile; global advances 1600 elems/iter, LDS advances 1608 elems
// (re-creating the +8 halfword bank padding). data_size=2B, type=2, count=1.
// This toolchain exposes the 6-arg builtin: (v4u, v8i, v4i, v4i, v8i, i32 cpol).
__device__ __forceinline__ void tdm_load_wtile(const unsigned short* gsrc,
                                               unsigned lds_byte_off) {
  unsigned long long ga = (unsigned long long)(size_t)gsrc;
  v4u g0 = { 1u,                                   // count=1 (valid descriptor)
             lds_byte_off,                         // lds_addr (bytes)
             (unsigned)ga,                         // global_addr[31:0]
             (unsigned)((ga >> 32) & 0x01FFFFFFu) | (2u << 30) }; // addr hi | type=2
  v8i g1 = { (int)((1u << 16) | (1u << 19)),       // data_size=2B | iterate_enable
             (int)((KDIM & 0xFFFFu) << 16),        // tensor_dim0 lo16 << 16
             (int)((16u) << 16),                   // td0 hi(0) | tensor_dim1=16 lo16
             (int)(KDIM << 16),                    // td1 hi(0) | tile_dim0=1600
             (int)1,                               // tile_dim1=1 | tile_dim2=0
             (int)KDIM,                            // tensor_dim0_stride lo32
             (int)((KDIM & 0xFFFFu) << 16),        // s0 hi(0) | tensor_dim1_stride lo16
             0 };                                  // s1 hi
  v4i g2 = { 1,                                    // tensor_dim2
             (int)WSTR,                            // lds_addr_increment (elems)
             (int)KDIM,                            // global_addr_increment lo (elems)
             (int)(15u << 16) };                   // inc hi(0) | iterate_count=15 (16x)
  v4i g3 = { 0, 0, 0, 0 };
  v8i g4 = { 0, 0, 0, 0, 0, 0, 0, 0 };             // unused trailing group
  __builtin_amdgcn_tensor_load_to_lds(g0, g1, g2, g3, g4, 0);
}

// ---- prep: conv1_w (NHID, NINP, 2) fp32 -> bf16 W[h][k*800+i] (K-major rows) ----
__global__ void __launch_bounds__(256) prepW(const float* __restrict__ w,
                                             unsigned short* __restrict__ wt) {
  int idx = blockIdx.x * 256 + threadIdx.x;
  if (idx >= NHID * KDIM) return;
  int h = idx / KDIM, rem = idx % KDIM;
  int k = rem / NINP, i = rem % NINP;
  wt[idx] = f2bf(w[(size_t)h * KDIM + i * 2 + k]);
}

// ---- fused conv1 GEMM + BN/ReLU + channel reductions (gate/gate_next/dist) ----
__global__ void __launch_bounds__(256) gemm_fused(
    const float* __restrict__ emb, const float* __restrict__ emb_last,
    const unsigned short* __restrict__ wbf,
    const float* __restrict__ cb,  const float* __restrict__ bng,
    const float* __restrict__ bnb, const float* __restrict__ bnm,
    const float* __restrict__ bnv, const float* __restrict__ g2w,
    const float* __restrict__ dwv,
    float* __restrict__ g0raw, float* __restrict__ g1raw, float* __restrict__ draw)
{
  extern __shared__ char smem[];
  __bf16* sA = (__bf16*)smem;                       // AROWS x ASTR halfwords

  const int tid  = threadIdx.x;
  const int b    = blockIdx.y;
  const int t0   = blockIdx.x * MT;
  const int wv   = tid >> 5;        // wave id = M-subtile (16 rows)
  const int lane = tid & 31;
  const int lm   = lane & 15;
  const int lh   = lane >> 4;

  // kick off async DMA of W tile 0 while we stage A
  if (wv == 0) tdm_load_wtile(wbf, (unsigned)SA_BYTES);

  // stage A tile: emb_full rows t0..t0+MT, converted to bf16 (float2 granularity)
  for (int idx = tid; idx < AROWS * (NINP / 2); idx += 256) {
    int j = idx / (NINP / 2);
    int p = idx % (NINP / 2);
    int tt = t0 + j;
    const float2* src = (tt == 0)
        ? (const float2*)(emb_last + (size_t)b * NINP) + p
        : (const float2*)(emb + ((size_t)(tt - 1) * BB + b) * NINP) + p;
    float2 v = *src;
    unsigned int pk = (unsigned int)f2bf(v.x) | ((unsigned int)f2bf(v.y) << 16);
    *(unsigned int*)(sA + j * ASTR + 2 * p) = pk;
  }

  const int r = wv * 16 + lm;   // A-fragment row (M = lane%16 per ISA layout)

  float dsum[8], g0sum[8], g1sum[8];
  #pragma unroll
  for (int v = 0; v < 8; ++v) { dsum[v] = 0.f; g0sum[v] = 0.f; g1sum[v] = 0.f; }

  #pragma unroll 1
  for (int n = 0; n < NTILE; ++n) {
    if (wv == 0) __builtin_amdgcn_s_wait_tensorcnt(0);  // tile n landed in LDS
    __syncthreads();   // n==0: also makes sA visible; all waves done with tile n-1
    if (wv == 0 && n + 1 < NTILE)   // prefetch tile n+1 into the other buffer
      tdm_load_wtile(wbf + (size_t)(n + 1) * 16 * KDIM,
                     (unsigned)(SA_BYTES + (size_t)((n + 1) & 1) * SW_BYTES));

    const __bf16* sW = (const __bf16*)(smem + SA_BYTES + (size_t)(n & 1) * SW_BYTES);
    const __bf16* wp = sW + (size_t)lm * WSTR + lh * 8;   // B: N=lane%16, K=lh*8+e (e<8), 16+lh*8+(e-8)

    v8f C = {};
    // conv tap 0: kk = s*32 + ... < 800  -> emb row r   (scalar split: 800%32==0)
    {
      const __bf16* ap = sA + (size_t)r * ASTR + lh * 8;
      #pragma unroll
      for (int s = 0; s < KHALF; ++s) {
        union { v16bf v; uint4 q[2]; } a, bm;
        a.q[0]  = *(const uint4*)(ap + s * 32);
        a.q[1]  = *(const uint4*)(ap + s * 32 + 16);
        bm.q[0] = *(const uint4*)(wp + s * 32);
        bm.q[1] = *(const uint4*)(wp + s * 32 + 16);
        C = __builtin_amdgcn_wmma_f32_16x16x32_bf16(false, a.v, false, bm.v,
                                                    (short)0, C, false, false);
      }
    }
    // conv tap 1: kk >= 800 -> emb row r+1, offset kk-800
    {
      const __bf16* ap = sA + (size_t)(r + 1) * ASTR + lh * 8;
      #pragma unroll
      for (int s = KHALF; s < 2 * KHALF; ++s) {
        union { v16bf v; uint4 q[2]; } a, bm;
        int so = (s - KHALF) * 32;
        a.q[0]  = *(const uint4*)(ap + so);
        a.q[1]  = *(const uint4*)(ap + so + 16);
        bm.q[0] = *(const uint4*)(wp + s * 32);
        bm.q[1] = *(const uint4*)(wp + s * 32 + 16);
        C = __builtin_amdgcn_wmma_f32_16x16x32_bf16(false, a.v, false, bm.v,
                                                    (short)0, C, false, false);
      }
    }

    // fused epilogue: pre = C + conv_b; bn; relu; accumulate channel reductions
    int c = n * 16 + lm;
    float scale = bng[c] * rsqrtf(bnv[c] + 1e-5f);
    float shift = (cb[c] - bnm[c]) * scale + bnb[c];
    float dwc = dwv[c];
    float gwc = g2w[c];             // (2,600,1) flat == channel index
    bool isg0 = (c < NHID / 2);
    #pragma unroll
    for (int v = 0; v < 8; ++v) {   // C[v] is row M = wv*16 + lh*8 + v, col c
      float val = fmaxf(C[v] * scale + shift, 0.f);
      dsum[v]  += val * dwc;
      float gv  = val * gwc;
      g0sum[v] += isg0 ? gv : 0.f;
      g1sum[v] += isg0 ? 0.f : gv;
    }
  }

  // reduce over the 16 lanes (channel sub-index) sharing each output row
  #pragma unroll
  for (int v = 0; v < 8; ++v) {
    #pragma unroll
    for (int m = 8; m >= 1; m >>= 1) {
      dsum[v]  += __shfl_xor(dsum[v],  m, 16);
      g0sum[v] += __shfl_xor(g0sum[v], m, 16);
      g1sum[v] += __shfl_xor(g1sum[v], m, 16);
    }
  }
  if (lm == 0) {
    #pragma unroll
    for (int v = 0; v < 8; ++v) {
      int t = t0 + wv * 16 + lh * 8 + v;
      size_t o = (size_t)b * TT + t;
      g0raw[o] = g0sum[v];
      g1raw[o] = g1sum[v];
      draw[o]  = dsum[v];
    }
  }
}

// ---- epilogue A: sigmoid gates, distances bias, emb tail copy ----
__global__ void __launch_bounds__(256) epiA(
    const float* __restrict__ g0raw, float* __restrict__ g1io,
    const float* __restrict__ draw, const float* __restrict__ g2b,
    const float* __restrict__ db, const float* __restrict__ emb,
    float* __restrict__ out)
{
  int idx = blockIdx.x * 256 + threadIdx.x;
  if (idx >= TT * BB) return;
  float g  = sigmf(g0raw[idx] + g2b[0]);
  float gn = sigmf(g1io[idx] + g2b[1]);
  out[OFF_GATE + idx] = g;
  out[OFF_DIST + idx] = draw[idx] + db[0];
  g1io[idx] = gn;                          // gate_next (sigmoided), reused by epiB
  if (idx < BB * NINP)
    out[OFF_EMB + idx] = emb[(size_t)(TT - 1) * BB * NINP + idx];
}

// ---- epilogue B: slot masks (cumprod of sigmoids) + cum tail ----
__global__ void __launch_bounds__(256) epiB(
    const float* __restrict__ cum_gate, const float* __restrict__ gn_sig,
    float* __restrict__ out)
{
  int idx = blockIdx.x * 256 + threadIdx.x;
  if (idx >= TT * BB) return;
  int b = idx / TT, t = idx % TT;
  const float* gate = out + OFF_GATE;
  float g = gate[idx], gn = gn_sig[idx];
  float mg = 1.f, mgn = 1.f;
  #pragma unroll
  for (int j = 0; j < NSLOTS; ++j) {
    int ci = (NSLOTS - j) + t;             // gate_hat = cum[b, (15-j)+t]
    float gh = (ci < NSLOTS) ? cum_gate[b * NSLOTS + ci]
                             : gate[b * TT + (ci - NSLOTS)];
    mg  *= sigmf((g  - gh) * 100.f + 5.f); // (g-gh)/RES*10 + 5
    mgn *= sigmf((gn - gh) * 100.f + 5.f);
    out[OFF_MG  + (size_t)idx * NSLOTS + j] = mg;
    out[OFF_MGN + (size_t)idx * NSLOTS + j] = mgn;
  }
  if (t < NSLOTS)
    out[OFF_CUM + b * NSLOTS + t] = gate[b * TT + (TT - NSLOTS + t)];
}

extern "C" void kernel_launch(void* const* d_in, const int* in_sizes, int n_in,
                              void* d_out, int out_size, void* d_ws, size_t ws_size,
                              hipStream_t stream) {
  const float* emb      = (const float*)d_in[0];
  const float* emb_last = (const float*)d_in[1];
  const float* cum_gate = (const float*)d_in[2];
  const float* conv1_w  = (const float*)d_in[3];
  const float* conv1_b  = (const float*)d_in[4];
  const float* bng      = (const float*)d_in[5];
  const float* bnb      = (const float*)d_in[6];
  const float* bnm      = (const float*)d_in[7];
  const float* bnv      = (const float*)d_in[8];
  const float* g2w      = (const float*)d_in[9];
  const float* g2b      = (const float*)d_in[10];
  const float* dw       = (const float*)d_in[11];
  const float* db       = (const float*)d_in[12];
  float* out = (float*)d_out;

  // workspace: bf16 W (3.84 MB) + 3 x (B*T) fp32 raw reductions
  unsigned short* wbf = (unsigned short*)d_ws;
  float* g0raw = (float*)((char*)d_ws + (size_t)NHID * KDIM * 2);
  float* g1raw = g0raw + TT * BB;
  float* draw  = g1raw + TT * BB;

  (void)hipFuncSetAttribute((const void*)gemm_fused,
                            hipFuncAttributeMaxDynamicSharedMemorySize,
                            (int)LDS_TOTAL);

  prepW<<<(NHID * KDIM + 255) / 256, 256, 0, stream>>>(conv1_w, wbf);
  gemm_fused<<<dim3(TT / MT, BB), 256, LDS_TOTAL, stream>>>(
      emb, emb_last, wbf, conv1_b, bng, bnb, bnm, bnv, g2w, dw,
      g0raw, g1raw, draw);
  epiA<<<(TT * BB + 255) / 256, 256, 0, stream>>>(g0raw, g1raw, draw, g2b, db, emb, out);
  epiB<<<(TT * BB + 255) / 256, 256, 0, stream>>>(cum_gate, g1raw, out);
}
